// Conv_19396072309398
// MI455X (gfx1250) — matrix operand order repaired and checked by reading, not compiled, not run on hardware
//
#include <hip/hip_runtime.h>
#include <hip/hip_bf16.h>
#include <math.h>

// ---------------------------------------------------------------------------
// SE(3)-conv fused kernel for MI455X (gfx1250, wave32, WMMA).
//
// Shapes: B=2, N=256, K=32, MID=128, FIBER_IN=FIBER_OUT=(32,16).
// Blocks (di,do): 00 (F=1024), 01 (F=512), 10 (F=512), 11 (F=768).
//
// Design: one workgroup = 4 nodes x 32 neighbors = 128 edges. The per-edge
// radial MLP (1->128 elementwise, LN, 128x128 GEMM, LN, 128xF GEMM) runs on
// v_wmma_f32_16x16x32_f16 with wave-private edge rows (no barriers in the
// MLP). R is staged per 64/48-column chunk in LDS (f16) and consumed
// immediately by the per-edge basis contraction; the masked mean over K uses
// ds_add_f32 LDS atomics. Weights are pre-transposed/converted to f16 into
// d_ws by a small prep kernel (needs ~832KB of workspace).
// ---------------------------------------------------------------------------

typedef _Float16 h16v __attribute__((ext_vector_type(16)));
typedef _Float16 h8v  __attribute__((ext_vector_type(8)));
typedef _Float16 h4v  __attribute__((ext_vector_type(4)));
typedef float    f8v  __attribute__((ext_vector_type(8)));

#define B_   2
#define N_   256
#define K_   32
#define NT_  4      // nodes per workgroup
#define SA_STRIDE 136   // halfs per A-stage row (272B, 16B aligned)
#define SR_STRIDE 72    // halfs per R-chunk row (144B, 16B aligned)

__device__ __forceinline__ float gelu_exact(float x) {
    return 0.5f * x * (1.0f + erff(x * 0.70710678118654752f));
}
__device__ __forceinline__ float wred32(float v) {
    v += __shfl_xor(v, 1, 32);  v += __shfl_xor(v, 2, 32);
    v += __shfl_xor(v, 4, 32);  v += __shfl_xor(v, 8, 32);
    v += __shfl_xor(v, 16, 32);
    return v;
}
__device__ __forceinline__ float hred16(float v) {   // reduce within half-wave
    v += __shfl_xor(v, 1, 32);  v += __shfl_xor(v, 2, 32);
    v += __shfl_xor(v, 4, 32);  v += __shfl_xor(v, 8, 32);
    return v;
}
__device__ __forceinline__ h16v cat8(h8v lo, h8v hi) {
    return __builtin_shufflevector(lo, hi, 0,1,2,3,4,5,6,7,8,9,10,11,12,13,14,15);
}
__device__ __forceinline__ f8v wmma16(h16v a, h16v b, f8v c) {
    // D = A(16x32 f16) x B(32x16 f16) + C(16x16 f32)
    return __builtin_amdgcn_wmma_f32_16x16x32_f16(false, a, false, b,
                                                  (short)0, c, false, false);
}

struct BlkW {
    const float *w0, *b0, *g0, *be0, *b1, *g1, *be1, *b2;
};

struct KParams {
    const float *x0, *x1, *rel, *b00, *b01, *b10, *b11, *si0, *si1;
    const int   *nbr, *msk;
    BlkW blk[4];
    const _Float16 *W1T[4];   // [128 out][128 in] f16, column(K)-contiguous
    const _Float16 *W2T[4];   // [F][128] f16, column(K)-contiguous
    float *out;
};

// ---------------------------------------------------------------------------
// One (di,do) block: radial MLP + immediate contraction.
//   BLK 0: (0,0) xg = x0g[32], basv = {basis00}        -> accA
//   BLK 1: (0,1) xg = x0g[32], basv = basis01[3]       -> accB
//   BLK 2: (1,0) xg = y[16] = sum_d basis10[d]*x1g[:,d]-> accA
//   BLK 3: (1,1) xg = x1g[48], basv = basis11[27]      -> accB
// ---------------------------------------------------------------------------
template<int F, int NCH, int BLK>
__device__ void run_block(const BlkW bp,
                          const _Float16* __restrict__ W1T,
                          const _Float16* __restrict__ W2T,
                          _Float16* sA, _Float16* sR, const float* sD,
                          const float* xg, const float* basv,
                          float* accA, float* accB)
{
    const int tid     = threadIdx.x;
    const int lane    = tid & 31;
    const int w       = tid >> 5;       // wave id 0..7 -> edge rows 16w..16w+15
    const int l16     = lane & 15;
    const int halfsel = lane >> 4;      // lane group within wave

    const f8v fzero = {0.f, 0.f, 0.f, 0.f, 0.f, 0.f, 0.f, 0.f};

    // ---- stage 1: a1 = LN(gelu(d * w0 + b0)), f16 into sA (wave-private rows)
    for (int m = 0; m < 16; ++m) {
        const int row = w * 16 + m;
        const float d = sD[row];
        const int c0 = lane * 4;
        float s0 = gelu_exact(d * bp.w0[c0 + 0] + bp.b0[c0 + 0]);
        float s1 = gelu_exact(d * bp.w0[c0 + 1] + bp.b0[c0 + 1]);
        float s2 = gelu_exact(d * bp.w0[c0 + 2] + bp.b0[c0 + 2]);
        float s3 = gelu_exact(d * bp.w0[c0 + 3] + bp.b0[c0 + 3]);
        float sum = wred32(s0 + s1 + s2 + s3);
        float sq  = wred32(s0 * s0 + s1 * s1 + s2 * s2 + s3 * s3);
        float mu   = sum * (1.0f / 128.0f);
        float rstd = rsqrtf(sq * (1.0f / 128.0f) - mu * mu + 1e-5f);
        h4v o;
        o[0] = (_Float16)((s0 - mu) * rstd * bp.g0[c0 + 0] + bp.be0[c0 + 0]);
        o[1] = (_Float16)((s1 - mu) * rstd * bp.g0[c0 + 1] + bp.be0[c0 + 1]);
        o[2] = (_Float16)((s2 - mu) * rstd * bp.g0[c0 + 2] + bp.be0[c0 + 2]);
        o[3] = (_Float16)((s3 - mu) * rstd * bp.g0[c0 + 3] + bp.be0[c0 + 3]);
        *(h4v*)&sA[row * SA_STRIDE + c0] = o;
    }

    // ---- GEMM1: S2 = A1 x W1  (wave-private M-tile, 8 N-tiles, K=128) ------
    const int rowA  = w * 16 + l16;
    const int koffA = halfsel * 8;      // A layout: lanes16-31 hold K+8 group
    h16v afr[4];
    for (int kt = 0; kt < 4; ++kt) {
        const _Float16* ap = &sA[rowA * SA_STRIDE + kt * 32 + koffA];
        afr[kt] = cat8(*(const h8v*)ap, *(const h8v*)(ap + 16));
    }
    f8v acc[8];
    for (int nt = 0; nt < 8; ++nt) acc[nt] = fzero;
    for (int kt = 0; kt < 4; ++kt) {
        for (int nt = 0; nt < 8; ++nt) {
            const _Float16* bptr = W1T + (nt * 16 + l16) * 128 + kt * 32 + halfsel * 8;
            h16v bfr = cat8(*(const h8v*)bptr, *(const h8v*)(bptr + 16));
            acc[nt] = wmma16(afr[kt], bfr, acc[nt]);
        }
    }
    // bias + gelu in place
    for (int nt = 0; nt < 8; ++nt) {
        const float b1v = bp.b1[nt * 16 + l16];
        for (int r = 0; r < 8; ++r) acc[nt][r] = gelu_exact(acc[nt][r] + b1v);
    }
    // per-row LN on accumulator layout (vgpr r -> row r / r+8), write A2 (f16)
    for (int r = 0; r < 8; ++r) {
        float sum = 0.f, sq = 0.f;
        for (int nt = 0; nt < 8; ++nt) { float v = acc[nt][r]; sum += v; sq += v * v; }
        sum = hred16(sum);  sq = hred16(sq);
        float mu   = sum * (1.0f / 128.0f);
        float rstd = rsqrtf(sq * (1.0f / 128.0f) - mu * mu + 1e-5f);
        const int row = w * 16 + r + halfsel * 8;
        for (int nt = 0; nt < 8; ++nt) {
            const int n = nt * 16 + l16;
            float v = (acc[nt][r] - mu) * rstd * bp.g1[n] + bp.be1[n];
            sA[row * SA_STRIDE + n] = (_Float16)v;
        }
    }
    // reload A fragments (A2); same-wave LDS ordering guarantees visibility
    for (int kt = 0; kt < 4; ++kt) {
        const _Float16* ap = &sA[rowA * SA_STRIDE + kt * 32 + koffA];
        afr[kt] = cat8(*(const h8v*)ap, *(const h8v*)(ap + 16));
    }

    // ---- GEMM2 in NCH-column chunks + fused contraction --------------------
    const int e  = tid >> 1;            // edge this thread contracts
    const int hh = tid & 1;
    constexpr int NTS    = NCH / 16;
    constexpr int NCHUNK = F / NCH;

    for (int ch = 0; ch < NCHUNK; ++ch) {
        f8v acc2[NTS];
        for (int nt = 0; nt < NTS; ++nt) acc2[nt] = fzero;
        for (int kt = 0; kt < 4; ++kt) {
            for (int nt = 0; nt < NTS; ++nt) {
                const _Float16* bptr = W2T + (ch * NCH + nt * 16 + l16) * 128
                                       + kt * 32 + halfsel * 8;
                h16v bfr = cat8(*(const h8v*)bptr, *(const h8v*)(bptr + 16));
                acc2[nt] = wmma16(afr[kt], bfr, acc2[nt]);
            }
        }
        for (int nt = 0; nt < NTS; ++nt) {
            const float b2v = bp.b2[ch * NCH + nt * 16 + l16];
            for (int r = 0; r < 8; ++r) {
                const int row = w * 16 + r + halfsel * 8;
                sR[row * SR_STRIDE + nt * 16 + l16] = (_Float16)(acc2[nt][r] + b2v);
            }
        }
        __syncthreads();

        if constexpr (BLK == 0) {            // F idx = o*32+i ; chunk -> o={2ch,2ch+1}
            const _Float16* rr = &sR[e * SR_STRIDE + hh * 32];
            float s = 0.f;
            for (int i = 0; i < 32; ++i) s += (float)rr[i] * xg[i];
            accA[ch] += basv[0] * s;         // o = 2*ch + hh
        } else if constexpr (BLK == 1) {     // F idx = o*32+i ; D=0..2
            const _Float16* rr = &sR[e * SR_STRIDE + hh * 32];
            float s = 0.f;
            for (int i = 0; i < 32; ++i) s += (float)rr[i] * xg[i];
            for (int D = 0; D < 3; ++D) accB[ch * 3 + D] += s * basv[D];
        } else if constexpr (BLK == 2) {     // F idx = o*16+i ; chunk -> 4 o's
            for (int oj = 0; oj < 2; ++oj) {
                const _Float16* rr = &sR[e * SR_STRIDE + (2 * oj + hh) * 16];
                float s = 0.f;
                for (int i = 0; i < 16; ++i) s += (float)rr[i] * xg[i]; // xg = y[16]
                accA[ch * 2 + oj] += s;      // o = 4ch + 2oj + hh
            }
        } else {                              // BLK==3: F idx = (o*16+i)*3+f ; chunk==o
            if ((ch & 1) == hh) {
                const _Float16* rr = &sR[e * SR_STRIDE];
                float aD0 = 0.f, aD1 = 0.f, aD2 = 0.f;
                for (int i = 0; i < 16; ++i) {
                    const float xv0 = xg[i * 3 + 0], xv1 = xg[i * 3 + 1], xv2 = xg[i * 3 + 2];
                    const float r0 = (float)rr[i * 3 + 0];
                    const float r1 = (float)rr[i * 3 + 1];
                    const float r2 = (float)rr[i * 3 + 2];
                    // t_D = sum_f r_f * sum_d basis11[D,d,f] * x1[i,d]
                    {
                        const float* bb = basv + 0 * 9;
                        aD0 += r0 * (bb[0] * xv0 + bb[3] * xv1 + bb[6] * xv2)
                             + r1 * (bb[1] * xv0 + bb[4] * xv1 + bb[7] * xv2)
                             + r2 * (bb[2] * xv0 + bb[5] * xv1 + bb[8] * xv2);
                    }
                    {
                        const float* bb = basv + 1 * 9;
                        aD1 += r0 * (bb[0] * xv0 + bb[3] * xv1 + bb[6] * xv2)
                             + r1 * (bb[1] * xv0 + bb[4] * xv1 + bb[7] * xv2)
                             + r2 * (bb[2] * xv0 + bb[5] * xv1 + bb[8] * xv2);
                    }
                    {
                        const float* bb = basv + 2 * 9;
                        aD2 += r0 * (bb[0] * xv0 + bb[3] * xv1 + bb[6] * xv2)
                             + r1 * (bb[1] * xv0 + bb[4] * xv1 + bb[7] * xv2)
                             + r2 * (bb[2] * xv0 + bb[5] * xv1 + bb[8] * xv2);
                    }
                }
                const int j = ch >> 1;
                accB[j * 3 + 0] += aD0;
                accB[j * 3 + 1] += aD1;
                accB[j * 3 + 2] += aD2;
            }
        }
        __syncthreads();
    }
}

// ---------------------------------------------------------------------------
__global__ __launch_bounds__(256) void conv_main(KParams P)
{
    __shared__ __align__(16) _Float16 sA[128 * SA_STRIDE];
    __shared__ __align__(16) _Float16 sR[128 * SR_STRIDE];
    __shared__ float sD[128];
    __shared__ float sMask[128];
    __shared__ float sAcc0[NT_][32];
    __shared__ float sAcc1[NT_][16][3];
    __shared__ int   sCnt[NT_];

    const int tid   = threadIdx.x;
    const int g     = blockIdx.x;
    const int b     = g >> 6;             // 64 node-tiles per batch
    const int nbase = (g & 63) * NT_;

    if (tid < NT_) sCnt[tid] = 0;
    for (int j = tid; j < NT_ * 32; j += 256) ((float*)sAcc0)[j] = 0.f;
    for (int j = tid; j < NT_ * 48; j += 256) ((float*)sAcc1)[j] = 0.f;
    if (tid < 128) {
        const int nl = tid >> 5, k = tid & 31;
        const int n  = nbase + nl;
        const long eidx = ((long)b * N_ + n) * K_ + k;
        sD[tid] = P.rel[eidx];
        const int m = P.msk[eidx];
        sMask[tid] = m ? 1.f : 0.f;
        if (m) atomicAdd(&sCnt[nl], 1);
    }
    __syncthreads();

    // per-thread edge for contraction
    const int e  = tid >> 1;
    const int hh = tid & 1;
    const int nl = e >> 5, k = e & 31;
    const int n  = nbase + nl;
    const long eidx = ((long)b * N_ + n) * K_ + k;
    const int  nb   = P.nbr[eidx];
    const long nbo0 = ((long)b * N_ + nb) * 32;
    const long nbo1 = ((long)b * N_ + nb) * 48;

    float accA[16], accB[24];
    for (int j = 0; j < 16; ++j) accA[j] = 0.f;
    for (int j = 0; j < 24; ++j) accB[j] = 0.f;

    {   // degree-0 input blocks (00, 01): gather x0 neighbors
        float x0g[32];
        for (int i = 0; i < 32; ++i) x0g[i] = P.x0[nbo0 + i];
        float bas0[1]; bas0[0] = P.b00[eidx];
        run_block<1024, 64, 0>(P.blk[0], P.W1T[0], P.W2T[0], sA, sR, sD,
                               x0g, bas0, accA, accB);
        float bas1[3];
        for (int D = 0; D < 3; ++D) bas1[D] = P.b01[eidx * 3 + D];
        run_block<512, 64, 1>(P.blk[1], P.W1T[1], P.W2T[1], sA, sR, sD,
                              x0g, bas1, accA, accB);
    }
    {   // degree-1 input blocks (10, 11): gather x1 neighbors
        float x1g[48];
        for (int i = 0; i < 48; ++i) x1g[i] = P.x1[nbo1 + i];
        float basd[3];
        for (int d = 0; d < 3; ++d) basd[d] = P.b10[eidx * 3 + d];
        float y[16];
        for (int i = 0; i < 16; ++i)
            y[i] = basd[0] * x1g[i * 3] + basd[1] * x1g[i * 3 + 1] + basd[2] * x1g[i * 3 + 2];
        run_block<512, 64, 2>(P.blk[2], P.W1T[2], P.W2T[2], sA, sR, sD,
                              y, basd, accA, accB);
        float bas11[27];
        for (int j = 0; j < 27; ++j) bas11[j] = P.b11[eidx * 27 + j];
        run_block<768, 48, 3>(P.blk[3], P.W1T[3], P.W2T[3], sA, sR, sD,
                              x1g, bas11, accA, accB);
    }

    // masked accumulation over neighbors (mean numerator) via LDS atomics
    const float mf = sMask[e];
    for (int j = 0; j < 16; ++j)
        atomicAdd(&sAcc0[nl][2 * j + hh], mf * accA[j]);
    for (int j = 0; j < 8; ++j)
        for (int D = 0; D < 3; ++D)
            atomicAdd(&sAcc1[nl][2 * j + hh][D], mf * accB[j * 3 + D]);
    __syncthreads();

    // masked mean + self-interaction residual, write outputs
    if (tid < NT_ * 32) {
        const int onl = tid >> 5, o = tid & 31;
        const int nn = nbase + onl;
        const float cnt = (float)sCnt[onl];
        float v = (cnt > 0.f) ? sAcc0[onl][o] / cnt : 0.f;
        const float* xs = P.x0 + ((long)b * N_ + nn) * 32;
        float si = 0.f;
        for (int d = 0; d < 32; ++d) si += xs[d] * P.si0[d * 32 + o];
        P.out[((long)b * N_ + nn) * 32 + o] = v + si;
    }
    for (int j = tid; j < NT_ * 48; j += 256) {
        const int onl = j / 48, r = j % 48, o = r / 3, m = r % 3;
        const int nn = nbase + onl;
        const float cnt = (float)sCnt[onl];
        float v = (cnt > 0.f) ? sAcc1[onl][o][m] / cnt : 0.f;
        const float* xs = P.x1 + ((long)b * N_ + nn) * 48;
        float si = 0.f;
        for (int d = 0; d < 16; ++d) si += xs[d * 3 + m] * P.si1[d * 16 + o];
        P.out[(long)B_ * N_ * 32 + (((long)b * N_ + nn) * 16 + o) * 3 + m] = v + si;
    }
}

// ---------------------------------------------------------------------------
// Prep: transpose + f16-convert W1 (4 x [128,128]) and W2 ([128,F]) into ws
// so WMMA B-fragments are K-contiguous ([out][in] layout).
// ---------------------------------------------------------------------------
struct PrepP {
    const float* w1[4];
    const float* w2[4];
    _Float16* ws;
};

__global__ __launch_bounds__(256) void prep_weights(PrepP P)
{
    int idx = blockIdx.x * 256 + threadIdx.x;
    const int W1TOT = 4 * 128 * 128;                     // 65536
    if (idx < W1TOT) {
        const int blk = idx >> 14;
        const int r   = idx & 16383;
        const int cp  = r >> 7, c = r & 127;             // out cp, in c
        P.ws[idx] = (_Float16)P.w1[blk][c * 128 + cp];
        return;
    }
    idx -= W1TOT;
    const int E0 = 1024 * 128, E1 = E0 + 512 * 128, E2 = E1 + 512 * 128, E3 = E2 + 768 * 128;
    if (idx >= E3) return;
    int blk, F, base;
    if      (idx < E0) { blk = 0; F = 1024; base = 0;  }
    else if (idx < E1) { blk = 1; F = 512;  base = E0; }
    else if (idx < E2) { blk = 2; F = 512;  base = E1; }
    else               { blk = 3; F = 768;  base = E2; }
    const int local = idx - base;
    const int f = local >> 7, c = local & 127;
    P.ws[W1TOT + base + local] = (_Float16)P.w2[blk][c * F + f];
}

// ---------------------------------------------------------------------------
extern "C" void kernel_launch(void* const* d_in, const int* in_sizes, int n_in,
                              void* d_out, int out_size, void* d_ws, size_t ws_size,
                              hipStream_t stream)
{
    (void)in_sizes; (void)n_in; (void)out_size; (void)ws_size;
    _Float16* wsh = (_Float16*)d_ws;

    // Input order (setup_inputs dict order, params flattened in insertion order):
    // 0:x0 1:x1 2:nbr 3:mask 4:rel 5:b00 6:b01 7:b10 8:b11
    // 9..48: blocks 00,01,10,11 x {w0,b0,g0,be0,w1,b1,g1,be1,w2,b2}; 49:si0 50:si1
    KParams kp;
    kp.x0  = (const float*)d_in[0];
    kp.x1  = (const float*)d_in[1];
    kp.nbr = (const int*)d_in[2];
    kp.msk = (const int*)d_in[3];
    kp.rel = (const float*)d_in[4];
    kp.b00 = (const float*)d_in[5];
    kp.b01 = (const float*)d_in[6];
    kp.b10 = (const float*)d_in[7];
    kp.b11 = (const float*)d_in[8];

    PrepP pp;
    int base = 9;
    for (int blk = 0; blk < 4; ++blk) {
        kp.blk[blk].w0  = (const float*)d_in[base + 0];
        kp.blk[blk].b0  = (const float*)d_in[base + 1];
        kp.blk[blk].g0  = (const float*)d_in[base + 2];
        kp.blk[blk].be0 = (const float*)d_in[base + 3];
        pp.w1[blk]      = (const float*)d_in[base + 4];
        kp.blk[blk].b1  = (const float*)d_in[base + 5];
        kp.blk[blk].g1  = (const float*)d_in[base + 6];
        kp.blk[blk].be1 = (const float*)d_in[base + 7];
        pp.w2[blk]      = (const float*)d_in[base + 8];
        kp.blk[blk].b2  = (const float*)d_in[base + 9];
        base += 10;
    }
    kp.si0 = (const float*)d_in[49];
    kp.si1 = (const float*)d_in[50];
    kp.out = (float*)d_out;
    pp.ws  = wsh;

    const size_t w2off[4] = {65536, 65536 + 131072, 65536 + 196608, 65536 + 262144};
    for (int blk = 0; blk < 4; ++blk) {
        kp.W1T[blk] = wsh + (size_t)blk * 16384;
        kp.W2T[blk] = wsh + w2off[blk];
    }

    const int prepElems = 65536 + 360448;                // 425984 f16 = 832KB ws
    prep_weights<<<(prepElems + 255) / 256, 256, 0, stream>>>(pp);
    conv_main<<<(B_ * N_) / NT_, 256, 0, stream>>>(kp);  // 128 workgroups x 8 waves
}
